// MSENet_80925773791533
// MI455X (gfx1250) — matrix operand order repaired and checked by reading, not compiled, not run on hardware
//
#include <hip/hip_runtime.h>
#include <cstdint>
#include <cstddef>

// ============================================================================
// MI455X (gfx1250, wave32) implementation.
//
// Roofline: after algebraic fusion (mean-pool pulled through the attention
// output: mean_i out = (1/N)(wv @ (x @ colsum(attn)) + N*bv)), the workload is
// ~12.5 GFLOP over ~140MB of activations (entire problem fits in the 192MB
// L2). All GEMMs (Q/K projections, energy Q^T K, wv@M) run on
// v_wmma_f32_16x16x32_f16 (f16 A/B, f32 accumulate). Softmax + reductions in
// f32 VALU. No atomics -> bitwise deterministic across replays.
//
// Data movement: k_proj_qk stages 128x16 slabs of x into LDS with the Tensor
// Data Mover (tensor_load_to_lds, TENSORcnt-tracked), double buffered so the
// DMA of slab s+1 overlaps the WMMAs of slab s. The D# tensor_dim0 OOB rule
// (reads past the tensor return zero) gives hardware zero-padding of the
// spatial dimension.
//
// WMMA wave32 fragment layouts (CDNA5 ISA 7.12.2):
//   A (16x32 f16): lane L row m=L&15;  half h -> K = (h<8?h:h+8) + ((L>>4)<<3)
//   B (32x16 f16): lane L col n=L&15;  half h -> K = (h<8?h:h+8) + ((L>>4)<<3)
//   D (16x16 f32): VGPR r -> m = r + ((L>>4)<<3), n = L&15
// ============================================================================

typedef __attribute__((ext_vector_type(16))) _Float16 v16h;
typedef __attribute__((ext_vector_type(8)))  float    v8f;
typedef __attribute__((ext_vector_type(4)))  unsigned int v4u;
typedef __attribute__((ext_vector_type(8)))  int      v8i;
typedef __attribute__((ext_vector_type(4)))  int      v4i;

#define N_WAYS   5
#define N_SHOTS  5
#define N_LEV    3
#define GAMMA_F  0.2f
#define SLAB_ROWS 128   // rows of x staged per TDM op (divides all C)

// ---------------------------------------------------------------------------
// TDM issue: DMA a [rows x 16] f32 tile (row stride = N floats) from global
// memory into LDS. Columns past (N - n0) read as zero (D# OOB rule).
// ---------------------------------------------------------------------------
__device__ __forceinline__ void tdm_load_slab(const float* gsrc, unsigned lds_byte_addr,
                                              int valid_cols, int rows, int row_stride)
{
    unsigned long long ga = (unsigned long long)(uintptr_t)gsrc;
    v4u g0;
    g0[0] = 1u;                                            // count=1, user mode
    g0[1] = lds_byte_addr;                                 // lds_addr
    g0[2] = (unsigned)(ga & 0xffffffffull);                // global_addr[31:0]
    g0[3] = (unsigned)((ga >> 32) & 0x01ffffffull)         // global_addr[56:32]
          | (2u << 30);                                    // type=2 ("image")
    v8i g1;
    g1[0] = (2 << 16);                                     // data_size = 4 bytes
    g1[1] = (int)(((unsigned)valid_cols & 0xffffu) << 16); // tensor_dim0[15:0]
    g1[2] = (int)((((unsigned)valid_cols >> 16) & 0xffffu) // tensor_dim0[31:16]
          | (((unsigned)rows & 0xffffu) << 16));           // tensor_dim1[15:0]
    g1[3] = (int)(16u << 16);                              // tile_dim0 = 16 units
    g1[4] = rows & 0xffff;                                 // tile_dim1 (tile_dim2=0)
    g1[5] = row_stride;                                    // tensor_dim0_stride[31:0]
    g1[6] = 0;                                             // stride hi / dim1_stride lo
    g1[7] = 0;
    v4i z = {0, 0, 0, 0};
#if defined(__clang_major__) && (__clang_major__ >= 23)
    v8i z8 = {0, 0, 0, 0, 0, 0, 0, 0};
    __builtin_amdgcn_tensor_load_to_lds(g0, g1, z, z, z8, 0);
#else
    __builtin_amdgcn_tensor_load_to_lds(g0, g1, z, z, 0);
#endif
}

// ---------------------------------------------------------------------------
// Kernel 1: Q/K projections.  One wave per 16x16 output tile of Q and K.
//   Q = wq(d x C) @ x(C x N) + bq ;  K = wk @ x + bk   (f16 out, N padded)
// x slabs arrive in LDS via double-buffered TDM; B-fragment reads are LDS.
// grid = (NPAD/16, d/16, Bg), block = 32
// ---------------------------------------------------------------------------
__global__ void k_proj_qk(const float* __restrict__ x,      // [Bg, C, N]
                          const float* __restrict__ wq, const float* __restrict__ bq,
                          const float* __restrict__ wk, const float* __restrict__ bk,
                          _Float16* __restrict__ Q,         // [Bg, d, NPAD]
                          _Float16* __restrict__ K,         // [Bg, d, NPAD]
                          int C, int N, int NPAD, int d)
{
    __shared__ float xs[2][SLAB_ROWS * 16];

    const int lane = threadIdx.x;            // 0..31
    const int n0   = blockIdx.x * 16;
    const int m0   = blockIdx.y * 16;
    const int b    = blockIdx.z;
    const float* xb = x + (size_t)b * C * N;

    // Flat LDS pointers carry the workgroup-relative byte offset in addr[31:0].
    const unsigned lds0      = (unsigned)(uintptr_t)&xs[0][0];
    const unsigned slabBytes = SLAB_ROWS * 16 * 4;
    const int      validCols = N - n0;       // >= 1 always (NPAD < N+16)
    const int      numSlabs  = C / SLAB_ROWS;

    const int mfr = lane & 15;               // A row / B col / D col
    const int hi8 = (lane >> 4) << 3;        // +8 for lanes 16..31

    v8f accQ = {};
    v8f accK = {};

    tdm_load_slab(xb + (size_t)0 * N + n0, lds0, validCols, SLAB_ROWS, N);

    for (int s = 0; s < numSlabs; ++s) {
        if (s + 1 < numSlabs) {
            tdm_load_slab(xb + (size_t)((s + 1) * SLAB_ROWS) * N + n0,
                          lds0 + (unsigned)((s + 1) & 1) * slabBytes,
                          validCols, SLAB_ROWS, N);
            __builtin_amdgcn_s_wait_tensorcnt(1);   // slab s landed, s+1 in flight
        } else {
            __builtin_amdgcn_s_wait_tensorcnt(0);
        }
        asm volatile("" ::: "memory");             // order LDS reads after the wait

        const float* slab = xs[s & 1];
        for (int kk = 0; kk < SLAB_ROWS; kk += 32) {
            const int k0 = s * SLAB_ROWS + kk;
            v16h aq, ak, bx;
#pragma unroll
            for (int h = 0; h < 16; ++h) {
                int k = k0 + (h < 8 ? h : h + 8) + hi8;
                aq[h] = (_Float16)wq[(m0 + mfr) * C + k];
                ak[h] = (_Float16)wk[(m0 + mfr) * C + k];
            }
            const int ko = kk + hi8;
#pragma unroll
            for (int h = 0; h < 16; ++h)
                bx[h] = (_Float16)slab[(ko + (h < 8 ? h : h + 8)) * 16 + mfr];
            accQ = __builtin_amdgcn_wmma_f32_16x16x32_f16(false, aq, false, bx,
                                                          (short)0, accQ, false, false);
            accK = __builtin_amdgcn_wmma_f32_16x16x32_f16(false, ak, false, bx,
                                                          (short)0, accK, false, false);
        }
    }

    const int n = n0 + mfr;
#pragma unroll
    for (int r = 0; r < 8; ++r) {
        int mm = m0 + r + hi8;
        Q[((size_t)b * d + mm) * NPAD + n] = (_Float16)(accQ[r] + bq[mm]);
        K[((size_t)b * d + mm) * NPAD + n] = (_Float16)(accK[r] + bk[mm]);
    }
}

// ---------------------------------------------------------------------------
// Kernel 2: energy = Q^T K for a 16-row strip, masked row softmax, column
// sums of the normalized strip -> cpart[b][strip][j]. 4 waves per block,
// wave w owns column tiles w, w+4, ...  E strip lives in LDS (<=50KB).
// grid = (NPAD/16, Bg), block = 128
// ---------------------------------------------------------------------------
template <int NPAD>
__global__ void k_attn(const _Float16* __restrict__ Q,   // [Bg, d, NPAD]
                       const _Float16* __restrict__ K,   // [Bg, d, NPAD]
                       float* __restrict__ cpart,        // [Bg, rowTiles, NPAD]
                       int N, int d, int rowTiles)
{
    __shared__ float E[16][NPAD];
    __shared__ float rowmax[16];
    __shared__ float rowsum[16];

    const int tid  = threadIdx.x;        // 0..127
    const int lane = tid & 31;
    const int wid  = tid >> 5;           // wave id 0..3
    const int i0   = blockIdx.x * 16;
    const int b    = blockIdx.y;
    const _Float16* Qb = Q + (size_t)b * d * NPAD;
    const _Float16* Kb = K + (size_t)b * d * NPAD;

    const int mfr = lane & 15;
    const int hi8 = (lane >> 4) << 3;
    const int jTiles = NPAD / 16;

    for (int jt = wid; jt < jTiles; jt += 4) {
        const int j0 = jt * 16;
        v8f acc = {};
        for (int k0 = 0; k0 < d; k0 += 32) {
            v16h af, bf;
#pragma unroll
            for (int h = 0; h < 16; ++h) {
                int k = k0 + (h < 8 ? h : h + 8) + hi8;
                af[h] = Qb[(size_t)k * NPAD + (i0 + mfr)];   // A[m][k] = Q[k][i0+m]
            }
            const int ko = k0 + hi8;
#pragma unroll
            for (int h = 0; h < 16; ++h) {
                bf[h] = Kb[(size_t)(ko + (h < 8 ? h : h + 8)) * NPAD + (j0 + mfr)];
            }
            acc = __builtin_amdgcn_wmma_f32_16x16x32_f16(false, af, false, bf,
                                                         (short)0, acc, false, false);
        }
        const int n = j0 + mfr;
#pragma unroll
        for (int r = 0; r < 8; ++r)
            E[r + hi8][n] = acc[r];
    }
    __syncthreads();

    // Row softmax stats: 8 threads per row (aligned 8-lane groups in a wave).
    const int row = tid >> 3;
    const int sub = tid & 7;
    float mx = -3.0e38f;
    for (int j = sub; j < NPAD; j += 8) {
        float e = (j < N) ? E[row][j] : -3.0e38f;   // mask padded columns
        E[row][j] = e;
        mx = fmaxf(mx, e);
    }
#pragma unroll
    for (int s = 4; s >= 1; s >>= 1) mx = fmaxf(mx, __shfl_xor(mx, s, 32));
    if (sub == 0) rowmax[row] = mx;
    __syncthreads();

    mx = rowmax[row];
    float sum = 0.0f;
    for (int j = sub; j < NPAD; j += 8) {
        float e = __expf(E[row][j] - mx);
        E[row][j] = e;
        sum += e;
    }
#pragma unroll
    for (int s = 4; s >= 1; s >>= 1) sum += __shfl_xor(sum, s, 32);
    if (sub == 0) rowsum[row] = sum;
    __syncthreads();

    // Deterministic partial column sums (only valid rows i0+r < N contribute).
    float* out = cpart + ((size_t)b * rowTiles + blockIdx.x) * NPAD;
    for (int j = tid; j < NPAD; j += blockDim.x) {
        float s = 0.0f;
        for (int r = 0; r < 16; ++r)
            if (i0 + r < N) s += E[r][j] / rowsum[r];
        out[j] = s;
    }
}

// ---------------------------------------------------------------------------
// Kernel 2b: a[b][j] = sum over row strips of cpart.
// ---------------------------------------------------------------------------
__global__ void k_colsum_reduce(const float* __restrict__ cpart,
                                float* __restrict__ a,
                                int rowTiles, int NPAD)
{
    const int b = blockIdx.y;
    const int j = blockIdx.x * blockDim.x + threadIdx.x;
    if (j >= NPAD) return;
    const float* p = cpart + (size_t)b * rowTiles * NPAD + j;
    float s = 0.0f;
    for (int t = 0; t < rowTiles; ++t) s += p[(size_t)t * NPAD];
    a[(size_t)b * NPAD + j] = s;
}

// ---------------------------------------------------------------------------
// Kernel 3a: m[b][c] = sum_j x[b,c,j]*a[b,j] (packed f16 as Mmat[c][bcol]),
//            xmean[b][c] = mean_j x. Padded b columns of Mmat zeroed.
// ---------------------------------------------------------------------------
__global__ void k_reduce_m(const float* __restrict__ x,     // [Bg, C, N]
                           const float* __restrict__ a,     // [Bg, NPAD]
                           _Float16* __restrict__ Mmat,     // [C, Bpad]
                           float* __restrict__ xmean,       // [Bg, C]
                           int C, int N, int NPAD, int Bg, int Bpad)
{
    const int idx = blockIdx.x * blockDim.x + threadIdx.x;
    if (idx >= Bpad * C) return;
    const int bcol = idx % Bpad;
    const int c    = idx / Bpad;
    if (bcol >= Bg) { Mmat[(size_t)c * Bpad + bcol] = (_Float16)0.0f; return; }
    const float* xr = x + ((size_t)bcol * C + c) * N;
    const float* ar = a + (size_t)bcol * NPAD;
    float mval = 0.0f, xs = 0.0f;
    for (int j = 0; j < N; ++j) {
        float xv = xr[j];
        xs   += xv;
        mval += xv * ar[j];
    }
    Mmat[(size_t)c * Bpad + bcol] = (_Float16)mval;
    xmean[(size_t)bcol * C + c]   = xs / (float)N;
}

// ---------------------------------------------------------------------------
// Kernel 3b: pooled = xmean + (gamma/N)*(wv @ M) + gamma*bv  via WMMA GEMM.
// grid = (Bpad/16, C/16), block = 32 (one wave per 16x16 tile).
// ---------------------------------------------------------------------------
__global__ void k_pool_gemm(const float* __restrict__ wvw,   // [C, C]
                            const float* __restrict__ bv,    // [C]
                            const _Float16* __restrict__ Mmat,// [C, Bpad]
                            const float* __restrict__ xmean, // [Bg, C]
                            float* __restrict__ pooled,      // [Bg, C]
                            int C, int N, int Bg, int Bpad)
{
    const int lane = threadIdx.x;
    const int n0   = blockIdx.x * 16;    // batch columns
    const int m0   = blockIdx.y * 16;    // channel rows
    const int mfr  = lane & 15;
    const int hi8  = (lane >> 4) << 3;

    v8f acc = {};
    for (int k0 = 0; k0 < C; k0 += 32) {
        v16h af, bf;
#pragma unroll
        for (int h = 0; h < 16; ++h) {
            int k = k0 + (h < 8 ? h : h + 8) + hi8;
            af[h] = (_Float16)wvw[(m0 + mfr) * C + k];
        }
        const int ko = k0 + hi8;
#pragma unroll
        for (int h = 0; h < 16; ++h)
            bf[h] = Mmat[(size_t)(ko + (h < 8 ? h : h + 8)) * Bpad + (n0 + mfr)];
        acc = __builtin_amdgcn_wmma_f32_16x16x32_f16(false, af, false, bf,
                                                     (short)0, acc, false, false);
    }
    const int bcol = n0 + mfr;
    if (bcol < Bg) {
        const float scale = GAMMA_F / (float)N;
#pragma unroll
        for (int r = 0; r < 8; ++r) {
            int c = m0 + r + hi8;
            pooled[(size_t)bcol * C + c] =
                xmean[(size_t)bcol * C + c] + scale * acc[r] + GAMMA_F * bv[c];
        }
    }
}

// ---------------------------------------------------------------------------
// Kernel 4a: class prototypes (mean of support features per label).
// ---------------------------------------------------------------------------
__global__ void k_protos(const float* __restrict__ pooled_s,  // [25, C]
                         const int* __restrict__ labels,      // [25]
                         float* __restrict__ proto,           // [WAYS, C]
                         int C)
{
    const int c = blockIdx.x * blockDim.x + threadIdx.x;
    if (c >= C) return;
    for (int w = 0; w < N_WAYS; ++w) {
        float acc = 0.0f, cnt = 0.0f;
        for (int s = 0; s < N_WAYS * N_SHOTS; ++s) {
            if (labels[s] == w) { acc += pooled_s[(size_t)s * C + c]; cnt += 1.0f; }
        }
        proto[(size_t)w * C + c] = acc / fmaxf(cnt, 1.0f);
    }
}

// ---------------------------------------------------------------------------
// Kernel 4b: logits[qi][w] += lw[l] * -sqrt(max(||qp - proto_w||^2, 0)).
// grid = 75 blocks (one per query), block = 128.
// ---------------------------------------------------------------------------
__global__ void k_logits(const float* __restrict__ pooled_q,  // [75, C]
                         const float* __restrict__ proto,     // [WAYS, C]
                         const float* __restrict__ lw, int lev,
                         float* __restrict__ out,              // [75, WAYS]
                         int C)
{
    __shared__ float red[N_WAYS * 128];
    const int qi = blockIdx.x;
    const int t  = threadIdx.x;
    float d2[N_WAYS] = {0.f, 0.f, 0.f, 0.f, 0.f};
    const float* qp = pooled_q + (size_t)qi * C;
    for (int c = t; c < C; c += 128) {
        float qv = qp[c];
#pragma unroll
        for (int w = 0; w < N_WAYS; ++w) {
            float diff = qv - proto[(size_t)w * C + c];
            d2[w] += diff * diff;
        }
    }
#pragma unroll
    for (int w = 0; w < N_WAYS; ++w) red[w * 128 + t] = d2[w];
    __syncthreads();
    for (int s = 64; s >= 1; s >>= 1) {
        if (t < s) {
#pragma unroll
            for (int w = 0; w < N_WAYS; ++w)
                red[w * 128 + t] += red[w * 128 + t + s];
        }
        __syncthreads();
    }
    if (t < N_WAYS) {
        float dd = fmaxf(red[t * 128], 0.0f);
        out[qi * N_WAYS + t] += lw[lev] * (-sqrtf(dd));
    }
}

// ===========================================================================
// Host orchestration.
// Inputs (setup_inputs order): for l in 0..2: s,q,wq,bq,wk,bk,wv,bv  then
// level_weights, support_labels, ways.
// ===========================================================================
extern "C" void kernel_launch(void* const* d_in, const int* in_sizes, int n_in,
                              void* d_out, int out_size, void* d_ws, size_t ws_size,
                              hipStream_t stream)
{
    static const int Cs[N_LEV] = {256, 512, 1024};
    static const int Ss[N_LEV] = {28, 14, 7};
    const int Bs = N_WAYS * N_SHOTS;   // 25
    const int Bq = 75;                 // WAYS*QPC

    const float* lw     = (const float*)d_in[8 * N_LEV + 0];
    const int*   labels = (const int*)  d_in[8 * N_LEV + 1];

    int Ns[N_LEV], NP[N_LEV], Dd[N_LEV], RT[N_LEV];
    for (int l = 0; l < N_LEV; ++l) {
        Ns[l] = Ss[l] * Ss[l];
        NP[l] = ((Ns[l] + 15) / 16) * 16;   // 784, 208, 64
        Dd[l] = Cs[l] / 8;                  // 32, 64, 128
        RT[l] = NP[l] / 16;
    }
    const int BpadQ = ((Bq + 15) / 16) * 16;   // 80
    const int BpadS = ((Bs + 15) / 16) * 16;   // 32

    // ---- workspace layout (~20.3MB, all fully overwritten each call) ------
    uintptr_t cur = (uintptr_t)d_ws;
    auto alloc = [&](size_t bytes) -> void* {
        void* p = (void*)cur;
        cur += (bytes + 255) & ~(size_t)255;
        return p;
    };
    float*    pooled_s[N_LEV]; float* pooled_q[N_LEV]; float* proto[N_LEV];
    for (int l = 0; l < N_LEV; ++l) {
        pooled_s[l] = (float*)alloc((size_t)Bs * Cs[l] * sizeof(float));
        pooled_q[l] = (float*)alloc((size_t)Bq * Cs[l] * sizeof(float));
        proto[l]    = (float*)alloc((size_t)N_WAYS * Cs[l] * sizeof(float));
    }
    size_t maxQK = 0, maxCp = 0, maxA = 0, maxM = 0, maxXm = 0;
    for (int l = 0; l < N_LEV; ++l) {
        size_t qk = (size_t)Bq * Dd[l] * NP[l];          if (qk > maxQK) maxQK = qk;
        size_t cp = (size_t)Bq * RT[l] * NP[l];          if (cp > maxCp) maxCp = cp;
        size_t av = (size_t)Bq * NP[l];                  if (av > maxA)  maxA  = av;
        size_t mm = (size_t)Cs[l] * BpadQ;               if (mm > maxM)  maxM  = mm;
        size_t xm = (size_t)Bq * Cs[l];                  if (xm > maxXm) maxXm = xm;
    }
    _Float16* Qws   = (_Float16*)alloc(maxQK * sizeof(_Float16));
    _Float16* Kws   = (_Float16*)alloc(maxQK * sizeof(_Float16));
    float*    cpart = (float*)   alloc(maxCp * sizeof(float));
    float*    avec  = (float*)   alloc(maxA  * sizeof(float));
    _Float16* Mmat  = (_Float16*)alloc(maxM  * sizeof(_Float16));
    float*    xmean = (float*)   alloc(maxXm * sizeof(float));
    (void)ws_size; (void)n_in; (void)in_sizes;

    hipMemsetAsync(d_out, 0, (size_t)out_size * sizeof(float), stream);

    for (int l = 0; l < N_LEV; ++l) {
        const int C = Cs[l], N = Ns[l], NPAD = NP[l], d = Dd[l], rowTiles = RT[l];
        const float* WQ = (const float*)d_in[8 * l + 2];
        const float* BQb= (const float*)d_in[8 * l + 3];
        const float* WK = (const float*)d_in[8 * l + 4];
        const float* BKb= (const float*)d_in[8 * l + 5];
        const float* WV = (const float*)d_in[8 * l + 6];
        const float* BV = (const float*)d_in[8 * l + 7];

        for (int g = 0; g < 2; ++g) {
            const float* x  = (const float*)d_in[8 * l + g];   // s (g=0) or q (g=1)
            const int Bg    = (g == 0) ? Bs : Bq;
            const int Bpad  = (g == 0) ? BpadS : BpadQ;
            float* pooled   = (g == 0) ? pooled_s[l] : pooled_q[l];

            k_proj_qk<<<dim3(NPAD / 16, d / 16, Bg), 32, 0, stream>>>(
                x, WQ, BQb, WK, BKb, Qws, Kws, C, N, NPAD, d);

            dim3 gAttn(rowTiles, Bg);
            if (NPAD == 784)
                k_attn<784><<<gAttn, 128, 0, stream>>>(Qws, Kws, cpart, N, d, rowTiles);
            else if (NPAD == 208)
                k_attn<208><<<gAttn, 128, 0, stream>>>(Qws, Kws, cpart, N, d, rowTiles);
            else
                k_attn<64><<<gAttn, 128, 0, stream>>>(Qws, Kws, cpart, N, d, rowTiles);

            k_colsum_reduce<<<dim3((NPAD + 255) / 256, Bg), 256, 0, stream>>>(
                cpart, avec, rowTiles, NPAD);

            int nThreads = Bpad * C;
            k_reduce_m<<<(nThreads + 255) / 256, 256, 0, stream>>>(
                x, avec, Mmat, xmean, C, N, NPAD, Bg, Bpad);

            k_pool_gemm<<<dim3(Bpad / 16, C / 16), 32, 0, stream>>>(
                WV, BV, Mmat, xmean, pooled, C, N, Bg, Bpad);
        }

        k_protos<<<(C + 255) / 256, 256, 0, stream>>>(pooled_s[l], labels, proto[l], C);
        k_logits<<<Bq, 128, 0, stream>>>(pooled_q[l], proto[l], lw, l, (float*)d_out, C);
    }
}
